// StdAttnBlock_15075335209395
// MI455X (gfx1250) — matrix operand order repaired and checked by reading, not compiled, not run on hardware
//
#include <hip/hip_runtime.h>
#include <math.h>

// ---------------------------------------------------------------------------
// Transformer block (pre-LN attn + MLP) for MI455X / gfx1250, wave32, WMMA.
// b=1, n=4096, d_model=1024, heads=16, d_head=64. fp32 in/out, bf16 matmuls
// with f32 accumulation via V_WMMA_F32_16X16X32_BF16.
// GEMMs: 128x64 block tile, K-step 64, double-buffered LDS, async A-tile copy.
// ---------------------------------------------------------------------------

#define DM   1024
#define NSEQ 4096
#define NH   16
#define DH   64

#define USE_ASYNC_LDS 1

typedef __attribute__((ext_vector_type(16))) __bf16 v16bf;
typedef __attribute__((ext_vector_type(8)))  float  v8f;

struct alignas(16) U4 { unsigned int x, y, z, w; };
struct U8 { U4 lo, hi; };

static __device__ inline unsigned short f2bfu(float f) {
  union { float f; unsigned int u; } v; v.f = f;
  unsigned int r = v.u + 0x7FFFu + ((v.u >> 16) & 1u); // round-to-nearest-even
  return (unsigned short)(r >> 16);
}

// Build a 16-element bf16 fragment (8 VGPRs) from two contiguous 16B chunks.
static __device__ inline v16bf load_frag(const unsigned short* p0,
                                         const unsigned short* p1) {
  U8 s;
  s.lo = *(const U4*)p0;
  s.hi = *(const U4*)p1;
  return __builtin_bit_cast(v16bf, s);
}

static __device__ inline v8f wmma_bf16(v16bf a, v16bf b, v8f c) {
  // D(f32 16x16) = A(16x32 bf16) * B(32x16 bf16) + C
  return __builtin_amdgcn_wmma_f32_16x16x32_bf16(false, a, false, b,
                                                 (short)0, c, false, false);
}

#if USE_ASYNC_LDS
// Per-lane 16B async copy global -> LDS (ASYNCcnt-tracked). IOFFSET applies to
// both the LDS and global addresses (CDNA5 ISA, GV mode).
static __device__ inline void async_cp16(unsigned lds_off,
                                         const unsigned short* gptr,
                                         int ioff) {
  asm volatile("global_load_async_to_lds_b128 %0, %1, off offset:%2"
               :: "v"(lds_off), "v"(gptr), "i"(ioff) : "memory");
}
static __device__ inline void async_wait0() {
  asm volatile("s_wait_asynccnt 0x0" ::: "memory");
}
#endif

// ---------------------------------------------------------------------------
// f32 -> bf16 conversion (weights)
// ---------------------------------------------------------------------------
__global__ void cvt_f32_bf16(const float* __restrict__ in,
                             unsigned short* __restrict__ out, int n) {
  int i = blockIdx.x * blockDim.x + threadIdx.x;
  int stride = gridDim.x * blockDim.x;
  for (; i < n; i += stride) out[i] = f2bfu(in[i]);
}

// ---------------------------------------------------------------------------
// LayerNorm: one block per row of 1024, fp32 math, bf16 output
// ---------------------------------------------------------------------------
__global__ __launch_bounds__(256) void ln_kernel(const float* __restrict__ x,
                                                 const float* __restrict__ w,
                                                 const float* __restrict__ b,
                                                 unsigned short* __restrict__ out) {
  __shared__ float sh1[256], sh2[256];
  int row = blockIdx.x;
  int tid = threadIdx.x;
  const float* xr = x + (size_t)row * DM;
  float s = 0.f, ss = 0.f;
  for (int i = tid; i < DM; i += 256) { float v = xr[i]; s += v; ss += v * v; }
  sh1[tid] = s; sh2[tid] = ss;
  __syncthreads();
  for (int o = 128; o > 0; o >>= 1) {
    if (tid < o) { sh1[tid] += sh1[tid + o]; sh2[tid] += sh2[tid + o]; }
    __syncthreads();
  }
  float mean = sh1[0] * (1.0f / DM);
  float var  = sh2[0] * (1.0f / DM) - mean * mean;
  float rstd = rsqrtf(var + 1e-5f);
  for (int i = tid; i < DM; i += 256)
    out[(size_t)row * DM + i] = f2bfu((xr[i] - mean) * rstd * w[i] + b[i]);
}

// ---------------------------------------------------------------------------
// Tiled WMMA GEMM: C(MxN) = A(MxK bf16) * B(KxN bf16) [+ epilogue]
// Block tile 128x64, K-step 64, double-buffered. 8 waves -> 4x2 grid of
// 32x32 wave tiles, 8 WMMAs per wave per K-step.
// ldsB holds the B tile pre-swizzled into B-fragment lane order so a lane's
// fragment is one contiguous 32B chunk. A-tile copied with async-to-LDS.
// ---------------------------------------------------------------------------
enum { EPI_BF16 = 0, EPI_RES_F32 = 1, EPI_BIAS_GELU_BF16 = 2, EPI_BIAS_RES_F32 = 3 };

template <int EPI>
__global__ __launch_bounds__(256) void gemm_wmma(const unsigned short* __restrict__ A,
                                                 const unsigned short* __restrict__ B,
                                                 const float* __restrict__ bias,
                                                 const float* __restrict__ resid,
                                                 void* __restrict__ Cout,
                                                 int M, int N, int K) {
  __shared__ alignas(32) unsigned short ldsA[2][128][64];          // row-major tiles
  __shared__ alignas(32) unsigned short ldsB[2][2][4][32][16];     // [buf][kchunk][colgrp][lane][elem]
  int tid  = threadIdx.x;
  int wave = tid >> 5, lane = tid & 31;
  int wm = wave & 3, wn = wave >> 2;
  int m0 = blockIdx.y * 128, n0 = blockIdx.x * 64;

  v8f acc[2][2];
#pragma unroll
  for (int t = 0; t < 2; t++)
#pragma unroll
    for (int g = 0; g < 2; g++)
#pragma unroll
      for (int r = 0; r < 8; r++) acc[t][g][r] = 0.f;

  int arow = tid >> 1, aseg = tid & 1;   // A: 128 rows x 2 segs of 32 (64B each)
  int bk = tid >> 2, bq = tid & 3;       // B: 64 k-rows x 4 n-groups of 16
  int bch = bk >> 5, bkk = bk & 31;
  int blane = (bkk >= 16) ? 16 : 0;
  int belem = bkk & 15;
  int lrow = lane & 15;
  int kofs = (lane < 16) ? 0 : 8;

  const unsigned short* gA = A + (size_t)(m0 + arow) * K + aseg * 32;
  const unsigned short* gB = B + (size_t)bk * N + n0 + bq * 16;
#if USE_ASYNC_LDS
  unsigned ldsA_off0 = (unsigned)(size_t)&ldsA[0][arow][aseg * 32];
  unsigned ldsA_off1 = (unsigned)(size_t)&ldsA[1][arow][aseg * 32];
#endif

  int NT = K >> 6;

  // ---- prologue: tile 0 into buffer 0 ----
  {
#if USE_ASYNC_LDS
#pragma unroll
    for (int i = 0; i < 4; i++) async_cp16(ldsA_off0, gA, i * 16);
#else
    const U4* ga = (const U4*)gA;
#pragma unroll
    for (int i = 0; i < 4; i++) ((U4*)&ldsA[0][arow][aseg * 32])[i] = ga[i];
#endif
    U4 bv0 = ((const U4*)gB)[0];
    U4 bv1 = ((const U4*)gB)[1];
    const unsigned short* bs0 = (const unsigned short*)&bv0;
    const unsigned short* bs1 = (const unsigned short*)&bv1;
#pragma unroll
    for (int j = 0; j < 8; j++) {
      ldsB[0][bch][bq][j + blane][belem]     = bs0[j];
      ldsB[0][bch][bq][j + 8 + blane][belem] = bs1[j];
    }
#if USE_ASYNC_LDS
    async_wait0();
#endif
    __syncthreads();
  }

  for (int kt = 0; kt < NT; kt++) {
    int buf = kt & 1;
    U4 bv0, bv1;
    bool more = (kt + 1) < NT;
    if (more) {
      // issue next tile's global traffic before compute (overlap with WMMA)
      int k1 = (kt + 1) << 6;
#if USE_ASYNC_LDS
      unsigned dsto = buf ? ldsA_off0 : ldsA_off1;
#pragma unroll
      for (int i = 0; i < 4; i++) async_cp16(dsto, gA + k1, i * 16);
#else
      const U4* ga = (const U4*)(gA + k1);
      U4 a0 = ga[0], a1 = ga[1], a2 = ga[2], a3 = ga[3];
      U4* da = (U4*)&ldsA[buf ^ 1][arow][aseg * 32];
      da[0] = a0; da[1] = a1; da[2] = a2; da[3] = a3;
#endif
      bv0 = ((const U4*)(gB + (size_t)k1 * N))[0];
      bv1 = ((const U4*)(gB + (size_t)k1 * N))[1];
    }

    // ---- compute: 2 k-chunks x (2x2) WMMAs on current buffer ----
#pragma unroll
    for (int ch = 0; ch < 2; ch++) {
      v16bf afr[2], bfr[2];
#pragma unroll
      for (int t = 0; t < 2; t++) {
        const unsigned short* pa = &ldsA[buf][wm * 32 + t * 16 + lrow][ch * 32];
        afr[t] = load_frag(pa + 2 * kofs, pa + 2 * kofs + 8);
      }
#pragma unroll
      for (int g = 0; g < 2; g++) {
        const unsigned short* pb = &ldsB[buf][ch][wn * 2 + g][lane][0];
        bfr[g] = load_frag(pb, pb + 8);
      }
#pragma unroll
      for (int t = 0; t < 2; t++)
#pragma unroll
        for (int g = 0; g < 2; g++)
          acc[t][g] = wmma_bf16(afr[t], bfr[g], acc[t][g]);
    }

    if (more) {
      // scatter staged B registers into the other buffer, then sync
      const unsigned short* bs0 = (const unsigned short*)&bv0;
      const unsigned short* bs1 = (const unsigned short*)&bv1;
#pragma unroll
      for (int j = 0; j < 8; j++) {
        ldsB[buf ^ 1][bch][bq][j + blane][belem]     = bs0[j];
        ldsB[buf ^ 1][bch][bq][j + 8 + blane][belem] = bs1[j];
      }
#if USE_ASYNC_LDS
      async_wait0();
#endif
      __syncthreads();
    }
  }

  // ---- epilogue ----
  int rofs = (lane < 16) ? 0 : 8;
#pragma unroll
  for (int t = 0; t < 2; t++)
#pragma unroll
    for (int g = 0; g < 2; g++)
#pragma unroll
      for (int r = 0; r < 8; r++) {
        int row = m0 + wm * 32 + t * 16 + rofs + r;
        int col = n0 + wn * 32 + g * 16 + lrow;
        size_t idx = (size_t)row * N + col;
        float v = acc[t][g][r];
        if (EPI == EPI_BF16) {
          ((unsigned short*)Cout)[idx] = f2bfu(v);
        } else if (EPI == EPI_RES_F32) {
          ((float*)Cout)[idx] = v + resid[idx];
        } else if (EPI == EPI_BIAS_GELU_BF16) {
          float h = v + bias[col];
          float gl = 0.5f * h * (1.0f + erff(h * 0.70710678118f));
          ((unsigned short*)Cout)[idx] = f2bfu(gl);
        } else {  // EPI_BIAS_RES_F32
          ((float*)Cout)[idx] = v + bias[col] + resid[idx];
        }
      }
}

// ---------------------------------------------------------------------------
// Causal flash attention. Grid: (qblock, head). 256 threads = 8 waves,
// each wave owns 16 query rows. Key blocks of 32. qkv layout: row-major
// [n][3*DM] with q at +0, k at +DM, v at +2*DM (head*DH within).
// ---------------------------------------------------------------------------
__global__ __launch_bounds__(256) void attn_kernel(const unsigned short* __restrict__ qkv,
                                                   unsigned short* __restrict__ attn) {
  __shared__ alignas(32) unsigned short ldsK[2][2][32][16]; // [dk-chunk][keygrp][lane][elem]
  __shared__ alignas(32) unsigned short ldsV[4][32][16];    // [dkgrp][lane][elem]
  __shared__ alignas(32) unsigned short ldsP[8][16][32];    // per-wave P scratch
  int tid  = threadIdx.x;
  int wave = tid >> 5, lane = tid & 31;
  int head = blockIdx.y;
  int qb   = blockIdx.x;
  int qbase  = qb * 128 + wave * 16;
  int lrow16 = lane & 15;
  int kofs = (lane < 16) ? 0 : 8;
  int rofs = (lane < 16) ? 0 : 8;

  // Q fragments resident in registers (16 rows x 64 dk = 2 chunks)
  v16bf aq[2];
  {
    const unsigned short* qp =
        qkv + (size_t)(qbase + lrow16) * (3 * DM) + head * DH;
#pragma unroll
    for (int ch = 0; ch < 2; ch++)
      aq[ch] = load_frag(qp + ch * 32 + 2 * kofs, qp + ch * 32 + 2 * kofs + 8);
  }

  v8f O[4];
#pragma unroll
  for (int g = 0; g < 4; g++)
#pragma unroll
    for (int r = 0; r < 8; r++) O[g][r] = 0.f;
  float mrow[8], lsum[8];
#pragma unroll
  for (int r = 0; r < 8; r++) { mrow[r] = -__builtin_inff(); lsum[r] = 0.f; }

  int ldkey = tid >> 3;  // 0..31
  int lddg  = tid & 7;   // dk group of 8
  int nkb = (qb + 1) * 4;  // uniform trip count across the block (causal mask)

  for (int kb = 0; kb < nkb; kb++) {
    int kbase = kb * 32;
    __syncthreads();
    // ---- K/V tiles -> LDS (B-fragment swizzled) ----
    {
      const unsigned short* kp =
          qkv + (size_t)(kbase + ldkey) * (3 * DM) + DM + head * DH + lddg * 8;
      U4 kv = *(const U4*)kp;
      const unsigned short* ks = (const unsigned short*)&kv;
#pragma unroll
      for (int j = 0; j < 8; j++) {
        int dk = lddg * 8 + j;
        int ch = dk >> 5, kk = dk & 31;
        ldsK[ch][ldkey >> 4][(ldkey & 15) + ((kk >= 16) ? 16 : 0)][kk & 15] = ks[j];
      }
      const unsigned short* vp =
          qkv + (size_t)(kbase + ldkey) * (3 * DM) + 2 * DM + head * DH + lddg * 8;
      U4 vv = *(const U4*)vp;
      const unsigned short* vs = (const unsigned short*)&vv;
#pragma unroll
      for (int j = 0; j < 8; j++) {
        int dk = lddg * 8 + j;
        ldsV[dk >> 4][(dk & 15) + ((ldkey >= 16) ? 16 : 0)][ldkey & 15] = vs[j];
      }
    }
    __syncthreads();

    // ---- S = Q * K^T (16x32 per wave, 4 WMMAs) ----
    v8f s0, s1;
#pragma unroll
    for (int r = 0; r < 8; r++) { s0[r] = 0.f; s1[r] = 0.f; }
#pragma unroll
    for (int ch = 0; ch < 2; ch++) {
      v16bf bk0 = load_frag(&ldsK[ch][0][lane][0], &ldsK[ch][0][lane][8]);
      v16bf bk1 = load_frag(&ldsK[ch][1][lane][0], &ldsK[ch][1][lane][8]);
      s0 = wmma_bf16(aq[ch], bk0, s0);
      s1 = wmma_bf16(aq[ch], bk1, s1);
    }

    // ---- online softmax (C-layout: VGPR r -> row rofs+r, col lane&15) ----
    float alpha[8];
#pragma unroll
    for (int r = 0; r < 8; r++) {
      int row  = qbase + rofs + r;
      int key0 = kbase + lrow16;
      int key1 = kbase + 16 + lrow16;
      float v0 = (key0 <= row) ? s0[r] * 0.125f : -__builtin_inff();
      float v1 = (key1 <= row) ? s1[r] * 0.125f : -__builtin_inff();
      float mx = fmaxf(v0, v1);
#pragma unroll
      for (int off = 1; off < 16; off <<= 1)
        mx = fmaxf(mx, __shfl_xor(mx, off, 32));
      float mN = fmaxf(mrow[r], mx);
      float a  = __expf(mrow[r] - mN);
      float p0 = __expf(v0 - mN);
      float p1 = __expf(v1 - mN);
      float rs = p0 + p1;
#pragma unroll
      for (int off = 1; off < 16; off <<= 1) rs += __shfl_xor(rs, off, 32);
      lsum[r] = lsum[r] * a + rs;
      mrow[r] = mN;
      alpha[r] = a;
      ldsP[wave][rofs + r][lrow16]      = f2bfu(p0);
      ldsP[wave][rofs + r][16 + lrow16] = f2bfu(p1);
    }
#pragma unroll
    for (int g = 0; g < 4; g++)
#pragma unroll
      for (int r = 0; r < 8; r++) O[g][r] *= alpha[r];

    // same-wave LDS RAW: DS ops are in-order per wave; fence the compiler+hw
    asm volatile("s_wait_dscnt 0" ::: "memory");

    // ---- O += P * V (4 WMMAs) ----
    {
      const unsigned short* pp = &ldsP[wave][lrow16][0];
      v16bf ap = load_frag(pp + 2 * kofs, pp + 2 * kofs + 8);
#pragma unroll
      for (int g = 0; g < 4; g++) {
        v16bf bvf = load_frag(&ldsV[g][lane][0], &ldsV[g][lane][8]);
        O[g] = wmma_bf16(ap, bvf, O[g]);
      }
    }
  }

  // ---- normalize + store bf16 ----
#pragma unroll
  for (int g = 0; g < 4; g++)
#pragma unroll
    for (int r = 0; r < 8; r++) {
      int row = qbase + rofs + r;
      int col = head * DH + g * 16 + lrow16;
      attn[(size_t)row * DM + col] = f2bfu(O[g][r] / lsum[r]);
    }
}

// ---------------------------------------------------------------------------
// Host-side orchestration
// ---------------------------------------------------------------------------
extern "C" void kernel_launch(void* const* d_in, const int* in_sizes, int n_in,
                              void* d_out, int out_size, void* d_ws, size_t ws_size,
                              hipStream_t stream) {
  (void)in_sizes; (void)n_in; (void)out_size; (void)ws_size;
  const float* x    = (const float*)d_in[0];
  const float* ln1w = (const float*)d_in[1];
  const float* ln1b = (const float*)d_in[2];
  const float* ln2w = (const float*)d_in[3];
  const float* ln2b = (const float*)d_in[4];
  const float* wqkv = (const float*)d_in[5];
  const float* wo   = (const float*)d_in[6];
  const float* wff1 = (const float*)d_in[7];
  const float* bff1 = (const float*)d_in[8];
  const float* wff2 = (const float*)d_in[9];
  const float* bff2 = (const float*)d_in[10];
  float* out = (float*)d_out;

  char* ws = (char*)d_ws;
  size_t off = 0;
  auto salloc = [&](size_t bytes) -> void* {
    void* p = ws + off;
    off += (bytes + 255) & ~(size_t)255;
    return p;
  };
  unsigned short* wqkv_bf = (unsigned short*)salloc((size_t)DM * 3 * DM * 2);
  unsigned short* wo_bf   = (unsigned short*)salloc((size_t)DM * DM * 2);
  unsigned short* wff1_bf = (unsigned short*)salloc((size_t)DM * 4 * DM * 2);
  unsigned short* wff2_bf = (unsigned short*)salloc((size_t)4 * DM * DM * 2);
  unsigned short* h_bf    = (unsigned short*)salloc((size_t)NSEQ * DM * 2);
  unsigned short* qkv_bf  = (unsigned short*)salloc((size_t)NSEQ * 3 * DM * 2);
  unsigned short* attn_bf = (unsigned short*)salloc((size_t)NSEQ * DM * 2);
  float*          x1      = (float*)salloc((size_t)NSEQ * DM * 4);
  unsigned short* h2_bf   = (unsigned short*)salloc((size_t)NSEQ * DM * 2);
  unsigned short* ff_bf   = (unsigned short*)salloc((size_t)NSEQ * 4 * DM * 2);

  // weight conversions
  cvt_f32_bf16<<<dim3(1024), dim3(256), 0, stream>>>(wqkv, wqkv_bf, DM * 3 * DM);
  cvt_f32_bf16<<<dim3(512),  dim3(256), 0, stream>>>(wo,   wo_bf,   DM * DM);
  cvt_f32_bf16<<<dim3(1024), dim3(256), 0, stream>>>(wff1, wff1_bf, DM * 4 * DM);
  cvt_f32_bf16<<<dim3(1024), dim3(256), 0, stream>>>(wff2, wff2_bf, 4 * DM * DM);

  // h = LN1(x)
  ln_kernel<<<dim3(NSEQ), dim3(256), 0, stream>>>(x, ln1w, ln1b, h_bf);

  // qkv = h @ w_qkv
  gemm_wmma<EPI_BF16><<<dim3(3 * DM / 64, NSEQ / 128), dim3(256), 0, stream>>>(
      h_bf, wqkv_bf, nullptr, nullptr, qkv_bf, NSEQ, 3 * DM, DM);

  // attention
  attn_kernel<<<dim3(NSEQ / 128, NH), dim3(256), 0, stream>>>(qkv_bf, attn_bf);

  // x1 = x + attn @ w_o
  gemm_wmma<EPI_RES_F32><<<dim3(DM / 64, NSEQ / 128), dim3(256), 0, stream>>>(
      attn_bf, wo_bf, nullptr, x, x1, NSEQ, DM, DM);

  // h2 = LN2(x1)
  ln_kernel<<<dim3(NSEQ), dim3(256), 0, stream>>>(x1, ln2w, ln2b, h2_bf);

  // ff = gelu(h2 @ w_ff1 + b_ff1)
  gemm_wmma<EPI_BIAS_GELU_BF16><<<dim3(4 * DM / 64, NSEQ / 128), dim3(256), 0, stream>>>(
      h2_bf, wff1_bf, bff1, nullptr, ff_bf, NSEQ, 4 * DM, DM);

  // out = x1 + ff @ w_ff2 + b_ff2
  gemm_wmma<EPI_BIAS_RES_F32><<<dim3(DM / 64, NSEQ / 128), dim3(256), 0, stream>>>(
      ff_bf, wff2_bf, bff2, x1, out, NSEQ, DM, 4 * DM);
}
